// StockPredictor_79723182949000
// MI455X (gfx1250) — matrix operand order repaired and checked by reading, not compiled, not run on hardware
//
#include <hip/hip_runtime.h>
#include <cstdint>
#include <cstddef>

// ---------------------------------------------------------------------------
// MI455X (gfx1250) implementation.
// Precompute phase: big f16 WMMA GEMMs (compute bound, ~250 GFLOP),
//   double-buffered via GLOBAL_LOAD_ASYNC_TO_LDS_B128 when available.
// Scan phase: persistent single-workgroup kernel (512 thr = 16 waves),
//   per-step WMMA GEMMs with L2-resident f16 weights, h kept in LDS.
// ---------------------------------------------------------------------------

typedef _Float16 half_t;
typedef _Float16 v16h __attribute__((ext_vector_type(16)));
typedef _Float16 v8h  __attribute__((ext_vector_type(8)));
typedef float    v8f  __attribute__((ext_vector_type(8)));

union ABr { v16h v; v8h h[2]; };

__device__ __forceinline__ v8f wmma16x16x32(const ABr& a, const ABr& b, v8f c) {
  return __builtin_amdgcn_wmma_f32_16x16x32_f16(false, a.v, false, b.v,
                                                (short)0, c, false, false);
}

__device__ __forceinline__ float sigm(float x) { return 1.0f / (1.0f + __expf(-x)); }

// ---- gfx1250 async global->LDS copy (ASYNCcnt) -----------------------------
#if defined(__has_builtin)
#if __has_builtin(__builtin_amdgcn_global_load_async_to_lds_b128)
#define HAVE_ASYNC_LDS 1
#endif
#endif

#ifdef HAVE_ASYNC_LDS
typedef int v4i __attribute__((ext_vector_type(4)));
typedef __attribute__((address_space(1))) v4i as1_v4i;   // global
typedef __attribute__((address_space(3))) v4i as3_v4i;   // LDS
__device__ __forceinline__ void async_b128(const half_t* g, half_t* l) {
  __builtin_amdgcn_global_load_async_to_lds_b128(
      (as1_v4i*)(void*)(half_t*)g, (as3_v4i*)(void*)l, 0, 0);
}
template <int N> __device__ __forceinline__ void wait_asynccnt() {
#if __has_builtin(__builtin_amdgcn_s_wait_asynccnt)
  __builtin_amdgcn_s_wait_asynccnt((short)N);
#else
  asm volatile("s_wait_asynccnt %0" :: "i"(N));
#endif
}
#define NSTAGE 2
#else
#define NSTAGE 1
#endif

// ---------------------------------------------------------------------------
// Generic precompute GEMM:  C[M,N] = act( rowscale[m] * (A[M,K] @ W[N,K]^T) + bias[n] )
// A, W are f16; writes f32 (outF) and/or f16 (outH).
// Block: 256 threads = 8 waves; block tile 128x128; wave tile 32x64; K-step 32.
// ---------------------------------------------------------------------------
__global__ __launch_bounds__(256)
void gemm_f16_wmma(const half_t* __restrict__ A, const half_t* __restrict__ W,
                   const float* __restrict__ bias, const float* __restrict__ rowscale,
                   float* __restrict__ outF, half_t* __restrict__ outH,
                   int M, int N, int K, int act)
{
  __shared__ half_t sA[NSTAGE][128][40];
  __shared__ half_t sB[NSTAGE][128][40];

  const int tid  = threadIdx.x;
  const int wave = tid >> 5;
  const int lane = tid & 31;
  const int lm   = lane & 15;
  const int ka   = (lane >> 4) << 3;   // A operand K sub-offset (0 or 8)
  const int kb   = (lane >> 4) << 4;   // unused (0 or 16); B operand uses ka too
  const int m0   = (wave >> 1) * 32;   // wave row slot (4 slots of 32)
  const int n0   = (wave & 1) * 64;    // wave col slot (2 slots of 64)

  const size_t blockM = (size_t)blockIdx.x * 128;
  const size_t blockN = (size_t)blockIdx.y * 128;

  const int lr = tid >> 1;             // LDS-fill row (0..127)
  const int lc = (tid & 1) * 16;       // LDS-fill col (0 or 16)
  const half_t* Ag = A + (blockM + (size_t)lr) * (size_t)K + lc;
  const half_t* Wg = W + (blockN + (size_t)lr) * (size_t)K + lc;

  const v8f vzero = {0.f,0.f,0.f,0.f,0.f,0.f,0.f,0.f};
  v8f acc[2][4];
#pragma unroll
  for (int i = 0; i < 2; ++i)
#pragma unroll
    for (int j = 0; j < 4; ++j) acc[i][j] = vzero;

#ifdef HAVE_ASYNC_LDS
  // Double-buffered pipeline: stage k0+32 while computing k0.
  auto issue = [&](int k0, int st) {
    async_b128(Ag + k0,     &sA[st][lr][lc]);
    async_b128(Ag + k0 + 8, &sA[st][lr][lc + 8]);
    async_b128(Wg + k0,     &sB[st][lr][lc]);
    async_b128(Wg + k0 + 8, &sB[st][lr][lc + 8]);
  };
  issue(0, 0);
  int stage = 0;
  for (int k0 = 0; k0 < K; k0 += 32, stage ^= 1) {
    const bool more = (k0 + 32) < K;
    if (more) { issue(k0 + 32, stage ^ 1); wait_asynccnt<4>(); }
    else      { wait_asynccnt<0>(); }
    __syncthreads();

    ABr a0, a1, bm;
    a0.h[0] = *(const v8h*)&sA[stage][m0 + lm][ka];
    a0.h[1] = *(const v8h*)&sA[stage][m0 + lm][ka + 16];
    a1.h[0] = *(const v8h*)&sA[stage][m0 + 16 + lm][ka];
    a1.h[1] = *(const v8h*)&sA[stage][m0 + 16 + lm][ka + 16];
#pragma unroll
    for (int j = 0; j < 4; ++j) {
      bm.h[0] = *(const v8h*)&sB[stage][n0 + j * 16 + lm][ka];
      bm.h[1] = *(const v8h*)&sB[stage][n0 + j * 16 + lm][ka + 16];
      acc[0][j] = wmma16x16x32(a0, bm, acc[0][j]);
      acc[1][j] = wmma16x16x32(a1, bm, acc[1][j]);
    }
    __syncthreads();   // buffer 'stage' may be overwritten next iteration
  }
#else
  for (int k0 = 0; k0 < K; k0 += 32) {
    __syncthreads();
    *(v8h*)&sA[0][lr][lc]     = *(const v8h*)(Ag + k0);
    *(v8h*)&sA[0][lr][lc + 8] = *(const v8h*)(Ag + k0 + 8);
    *(v8h*)&sB[0][lr][lc]     = *(const v8h*)(Wg + k0);
    *(v8h*)&sB[0][lr][lc + 8] = *(const v8h*)(Wg + k0 + 8);
    __syncthreads();

    ABr a0, a1, bm;
    a0.h[0] = *(const v8h*)&sA[0][m0 + lm][ka];
    a0.h[1] = *(const v8h*)&sA[0][m0 + lm][ka + 16];
    a1.h[0] = *(const v8h*)&sA[0][m0 + 16 + lm][ka];
    a1.h[1] = *(const v8h*)&sA[0][m0 + 16 + lm][ka + 16];
#pragma unroll
    for (int j = 0; j < 4; ++j) {
      bm.h[0] = *(const v8h*)&sB[0][n0 + j * 16 + lm][ka];
      bm.h[1] = *(const v8h*)&sB[0][n0 + j * 16 + lm][ka + 16];
      acc[0][j] = wmma16x16x32(a0, bm, acc[0][j]);
      acc[1][j] = wmma16x16x32(a1, bm, acc[1][j]);
    }
  }
#endif

#pragma unroll
  for (int mt = 0; mt < 2; ++mt)
#pragma unroll
    for (int nt = 0; nt < 4; ++nt) {
      const size_t gcol = blockN + (size_t)(n0 + nt * 16 + lm);
#pragma unroll
      for (int r = 0; r < 8; ++r) {
        const size_t grow = blockM + (size_t)(m0 + mt * 16 + ka + r);
        float v = acc[mt][nt][r];
        if (rowscale) v *= rowscale[grow];
        if (bias)     v += bias[gcol];
        if (act == 1) v = fmaxf(v, 0.f);
        const size_t o = grow * (size_t)N + gcol;
        if (outF) outF[o] = v;
        if (outH) outH[o] = (half_t)v;
      }
    }
}

// ---------------------------------------------------------------------------
// Small helper kernels
// ---------------------------------------------------------------------------
__global__ void cvt_kernel(const float* __restrict__ in, half_t* __restrict__ out, int n) {
  int i = blockIdx.x * blockDim.x + threadIdx.x;
  if (i < n) out[i] = (half_t)in[i];
}

__global__ void rowscale_kernel(const float* __restrict__ td, float* __restrict__ rs, int n) {
  int i = blockIdx.x * blockDim.x + threadIdx.x;
  if (i < n) rs[i] = __expf(-0.1f * td[i]);
}

__global__ void gather_kernel(const int* __restrict__ evs, const float* __restrict__ table,
                              half_t* __restrict__ out) {
  int i = blockIdx.x * blockDim.x + threadIdx.x;
  if (i < 131072 * 128) {
    int d = i & 127;
    int r = i >> 7;
    out[i] = (half_t)table[(size_t)evs[r] * 128 + d];
  }
}

// Laplacian 3x3 over (T,F) with SAME zero padding, output f16.
__global__ void lap_kernel(const float* __restrict__ x, half_t* __restrict__ out) {
  int i = blockIdx.x * blockDim.x + threadIdx.x;
  if (i >= 32 * 512 * 128) return;
  int f = i & 127;
  int t = (i >> 7) & 511;
  int b = i >> 16;
  const float* xb = x + (size_t)b * 512 * 128;
  float c  = xb[t * 128 + f];
  float up = (t > 0)   ? xb[(t - 1) * 128 + f] : 0.f;
  float dn = (t < 511) ? xb[(t + 1) * 128 + f] : 0.f;
  float lf = (f > 0)   ? xb[t * 128 + f - 1]   : 0.f;
  float rt = (f < 127) ? xb[t * 128 + f + 1]   : 0.f;
  float ul = (t > 0 && f > 0)     ? xb[(t - 1) * 128 + f - 1] : 0.f;
  float ur = (t > 0 && f < 127)   ? xb[(t - 1) * 128 + f + 1] : 0.f;
  float dl = (t < 511 && f > 0)   ? xb[(t + 1) * 128 + f - 1] : 0.f;
  float dr = (t < 511 && f < 127) ? xb[(t + 1) * 128 + f + 1] : 0.f;
  out[i] = (half_t)(4.f * c - 2.f * (up + dn + lf + rt) + ul + ur + dl + dr);
}

// ---------------------------------------------------------------------------
// Per-step GEMM used inside the scan. M=32 fixed (2 m-tiles). A can be split:
// k < kSplit reads A1 (usually LDS), k >= kSplit reads A2 (f16, any space).
// Waves split the N/16 tiles round-robin (stride = blockDim/32).
// ---------------------------------------------------------------------------
__device__ void scan_gemm(const half_t* A1, int lda1,
                          const half_t* A2, size_t lda2, int kSplit,
                          int K, const half_t* __restrict__ W,
                          const float* __restrict__ bias,
                          const float* __restrict__ add, size_t addStride,
                          float* __restrict__ Cout, int N, int act)
{
  const int tid  = threadIdx.x;
  const int wave = tid >> 5;
  const int nw   = blockDim.x >> 5;
  const int lane = tid & 31;
  const int lm   = lane & 15;
  const int ka   = (lane >> 4) << 3;
  const int kb   = (lane >> 4) << 4;
  const int ntiles = N >> 4;
  const v8f vzero = {0.f,0.f,0.f,0.f,0.f,0.f,0.f,0.f};

  for (int nt = wave; nt < ntiles; nt += nw) {
    v8f acc0 = vzero, acc1 = vzero;
    const half_t* wrow = W + (size_t)(nt * 16 + lm) * (size_t)K;
    if (nt + nw < ntiles)   // prefetch next weight tile row (L2 streaming)
      __builtin_prefetch(W + (size_t)((nt + nw) * 16 + lm) * (size_t)K, 0, 1);
    for (int k = 0; k < K; k += 32) {
      ABr a0, a1, bm;
      if (k < kSplit) {
        const half_t* p0 = A1 + (size_t)lm * lda1 + k + ka;
        const half_t* p1 = A1 + (size_t)(lm + 16) * lda1 + k + ka;
        a0.h[0] = *(const v8h*)p0;  a0.h[1] = *(const v8h*)(p0 + 16);
        a1.h[0] = *(const v8h*)p1;  a1.h[1] = *(const v8h*)(p1 + 16);
      } else {
        const int kk = k - kSplit;
        const half_t* p0 = A2 + (size_t)lm * lda2 + kk + ka;
        const half_t* p1 = A2 + (size_t)(lm + 16) * lda2 + kk + ka;
        a0.h[0] = *(const v8h*)p0;  a0.h[1] = *(const v8h*)(p0 + 16);
        a1.h[0] = *(const v8h*)p1;  a1.h[1] = *(const v8h*)(p1 + 16);
      }
      const half_t* pb = wrow + k + ka;
      bm.h[0] = *(const v8h*)pb;  bm.h[1] = *(const v8h*)(pb + 16);
      acc0 = wmma16x16x32(a0, bm, acc0);
      acc1 = wmma16x16x32(a1, bm, acc1);
    }
    const int col = nt * 16 + lm;
    const float bc = bias ? bias[col] : 0.0f;
#pragma unroll
    for (int r = 0; r < 8; ++r) {
      const int row0 = ka + r;
      const int row1 = 16 + ka + r;
      float v0 = acc0[r] + bc;
      float v1 = acc1[r] + bc;
      if (add) {
        v0 += add[(size_t)row0 * addStride + col];
        v1 += add[(size_t)row1 * addStride + col];
      }
      if (act == 1)      { v0 = fmaxf(v0, 0.f); v1 = fmaxf(v1, 0.f); }
      else if (act == 2) { v0 = sigm(v0);       v1 = sigm(v1);       }
      Cout[(size_t)row0 * N + col] = v0;
      Cout[(size_t)row1 * N + col] = v1;
    }
  }
}

// ---------------------------------------------------------------------------
// Persistent scan kernel: one workgroup, 512 threads (16 waves), 512 steps.
// h kept as f16 in LDS (32 x 512).
// ---------------------------------------------------------------------------
__global__ __launch_bounds__(512)
void scan_kernel(const half_t* __restrict__ wq16, const float* __restrict__ bq,
                 const half_t* __restrict__ whh16, const float* __restrict__ bhh,
                 const float* __restrict__ gatesx,
                 const half_t* __restrict__ k16, const half_t* __restrict__ v16,
                 const half_t* __restrict__ wout16, const float* __restrict__ bout,
                 const half_t* __restrict__ wig16, const float* __restrict__ big,
                 const half_t* __restrict__ weg16, const float* __restrict__ beg,
                 const half_t* __restrict__ wp116, const float* __restrict__ bp1,
                 const float* __restrict__ wp2, const float* __restrict__ bp2,
                 const half_t* __restrict__ ih16, const float* __restrict__ ih32,
                 float* q_ws, float* gates_ws, float* h0_ws, float* gi_ws,
                 float* ev_ws, float* h1_ws, float* ge_ws, float* p1_ws, float* c_ws,
                 half_t* x016g, half_t* y16g,
                 float* __restrict__ out)
{
  __shared__ half_t h16[32 * 512];
  const int tid = threadIdx.x;
  const int nthr = blockDim.x;

  for (int i = tid; i < 32 * 512; i += nthr) { h16[i] = (half_t)0.f; c_ws[i] = 0.f; }
  __threadfence();
  __syncthreads();

  for (int t = 0; t < 512; ++t) {
    // Phase A: q = h@wq^T + bq ; gates = gates_x[t] + h@w_hh^T + b_hh
    scan_gemm(h16, 512, nullptr, 0, 512, 512, wq16, bq, nullptr, 0, q_ws, 512, 0);
    scan_gemm(h16, 512, nullptr, 0, 512, 512, whh16, bhh,
              gatesx + (size_t)t * 2048, (size_t)512 * 2048, gates_ws, 2048, 0);
    __threadfence(); __syncthreads();

    // Phase B: LSTM pointwise (c1, h0) + event attention (q against k/v at t)
    for (int i = tid; i < 32 * 512; i += nthr) {
      int b = i >> 9, j = i & 511;
      const float* g = gates_ws + (size_t)b * 2048;
      float ig = sigm(g[j]);
      float fg = sigm(g[512 + j]);
      float gg = tanhf(g[1024 + j]);
      float og = sigm(g[1536 + j]);
      float c1 = fg * c_ws[i] + ig * gg;
      c_ws[i] = c1;
      float h0 = og * tanhf(c1);
      h0_ws[i] = h0;
      x016g[i] = (half_t)h0;
    }
    if (tid < 128) {
      int b = tid >> 2, hd = tid & 3;
      const float* q = q_ws + (size_t)b * 512 + hd * 128;
      const half_t* kb = k16 + ((size_t)(b * 512 + t) * 8) * 512 + hd * 128;
      const half_t* vb = v16 + ((size_t)(b * 512 + t) * 8) * 512 + hd * 128;
      float s[8], mx = -1e30f;
      for (int e = 0; e < 8; ++e) {
        const half_t* kr = kb + (size_t)e * 512;
        float d = 0.f;
        for (int dd = 0; dd < 128; ++dd) d += q[dd] * (float)kr[dd];
        s[e] = d * 0.08838834764831845f;
        mx = fmaxf(mx, s[e]);
      }
      float den = 0.f;
      for (int e = 0; e < 8; ++e) { s[e] = __expf(s[e] - mx); den += s[e]; }
      float inv = 1.f / den;
      for (int dd = 0; dd < 128; ++dd) {
        float o = 0.f;
        for (int e = 0; e < 8; ++e) o += s[e] * (float)vb[(size_t)e * 512 + dd];
        y16g[(size_t)b * 512 + hd * 128 + dd] = (half_t)(o * inv);
      }
    }
    __threadfence(); __syncthreads();

    // Phase C: ev = attn@w_out^T + b_out ; gi = sigmoid([h0|ih_t]@w_ig^T + b_ig)
    scan_gemm(y16g, 512, nullptr, 0, 512, 512, wout16, bout, nullptr, 0, ev_ws, 512, 0);
    scan_gemm(x016g, 512, ih16 + (size_t)t * 512, (size_t)512 * 512, 512, 1024,
              wig16, big, nullptr, 0, gi_ws, 512, 2);
    __threadfence(); __syncthreads();

    // Phase D: h1 = h0*gi + ih*(1-gi); stage h1,ev as f16
    for (int i = tid; i < 32 * 512; i += nthr) {
      int b = i >> 9, j = i & 511;
      float ih = ih32[((size_t)b * 512 + t) * 512 + j];
      float gi = gi_ws[i];
      float h1 = h0_ws[i] * gi + ih * (1.f - gi);
      h1_ws[i] = h1;
      x016g[i] = (half_t)h1;
      y16g[i] = (half_t)ev_ws[i];
    }
    __threadfence(); __syncthreads();

    // Phase E: ge = sigmoid([h1|ev]@w_eg^T + b_eg)
    scan_gemm(x016g, 512, y16g, 512, 512, 1024, weg16, beg, nullptr, 0, ge_ws, 512, 2);
    __threadfence(); __syncthreads();

    // Phase F: h2 = h1*ge + ev*(1-ge) -> becomes h for next step (LDS f16)
    for (int i = tid; i < 32 * 512; i += nthr) {
      float ge = ge_ws[i];
      float h2 = h1_ws[i] * ge + ev_ws[i] * (1.f - ge);
      h16[i] = (half_t)h2;
    }
    __threadfence(); __syncthreads();

    // Phase G: p1 = relu(h2@w_p1^T + b_p1); pred = p1@w_p2^T + b_p2
    scan_gemm(h16, 512, nullptr, 0, 512, 512, wp116, bp1, nullptr, 0, p1_ws, 256, 1);
    __threadfence(); __syncthreads();
    if (tid < 32) {
      const float* p = p1_ws + (size_t)tid * 256;
      float sacc = bp2[0];
      for (int i2 = 0; i2 < 256; ++i2) sacc += p[i2] * wp2[i2];
      out[(size_t)tid * 512 + t] = sacc;
    }
    __threadfence(); __syncthreads();
  }
}

// ---------------------------------------------------------------------------
// Host side
// ---------------------------------------------------------------------------
extern "C" void kernel_launch(void* const* d_in, const int* in_sizes, int n_in,
                              void* d_out, int out_size, void* d_ws, size_t ws_size,
                              hipStream_t stream)
{
  (void)in_sizes; (void)out_size; (void)ws_size;
  if (n_in < 50) return;

  // Input leaf order: x, events, time_distances, then params depth-first.
  const float* x       = (const float*)d_in[0];
  const int*   evs     = (const int*)d_in[1];
  const float* td      = (const float*)d_in[2];
  // params['groups'] (indices 3..26) are dead code in the reference.
  const float* embT    = (const float*)d_in[27];
  const float* imp_w1  = (const float*)d_in[28];
  const float* imp_b1  = (const float*)d_in[29];
  const float* imp_w2  = (const float*)d_in[30];
  const float* imp_b2  = (const float*)d_in[31];
  const float* ev_w_in = (const float*)d_in[32];
  const float* ev_b_in = (const float*)d_in[33];
  const float* ev_w_out= (const float*)d_in[34];
  const float* ev_b_out= (const float*)d_in[35];
  const float* w_ih    = (const float*)d_in[36];
  const float* b_ih    = (const float*)d_in[37];
  const float* w_hh    = (const float*)d_in[38];
  const float* b_hh    = (const float*)d_in[39];
  const float* w_ip    = (const float*)d_in[40];
  const float* b_ip    = (const float*)d_in[41];
  const float* w_ig    = (const float*)d_in[42];
  const float* b_ig    = (const float*)d_in[43];
  const float* w_eg    = (const float*)d_in[44];
  const float* b_eg    = (const float*)d_in[45];
  const float* w_p1    = (const float*)d_in[46];
  const float* b_p1    = (const float*)d_in[47];
  const float* w_p2    = (const float*)d_in[48];
  const float* b_p2    = (const float*)d_in[49];

  uint8_t* wsp = (uint8_t*)d_ws;
  size_t off = 0;
  auto ALLOC = [&](size_t bytes) -> void* {
    void* p = wsp + off;
    off += (bytes + 255) & ~(size_t)255;
    return p;
  };

  const int BT  = 32 * 512;      // 16384 (b,t) rows
  const int BTN = BT * 8;        // 131072 (b,t,e) rows

  half_t* x16       = (half_t*)ALLOC((size_t)BT * 128 * 2);
  half_t* lap16     = (half_t*)ALLOC((size_t)BT * 128 * 2);
  half_t* emb16     = (half_t*)ALLOC((size_t)BTN * 128 * 2);
  half_t* imp1_16   = (half_t*)ALLOC((size_t)BTN * 512 * 2);
  half_t* imp2_16   = (half_t*)ALLOC((size_t)BTN * 512 * 2);   // unscaled (bias applied)
  half_t* k16       = (half_t*)ALLOC((size_t)BTN * 512 * 2);
  half_t* v16       = (half_t*)ALLOC((size_t)BTN * 512 * 2);
  float*  rs        = (float*) ALLOC((size_t)BTN * 4);
  float*  ih32      = (float*) ALLOC((size_t)BT * 512 * 4);
  half_t* ih16      = (half_t*)ALLOC((size_t)BT * 512 * 2);
  float*  gatesx    = (float*) ALLOC((size_t)BT * 2048 * 4);
  half_t* w_ih16    = (half_t*)ALLOC((size_t)2048 * 128 * 2);
  half_t* w_hh16    = (half_t*)ALLOC((size_t)2048 * 512 * 2);
  half_t* imp_w1_16 = (half_t*)ALLOC((size_t)512 * 128 * 2);
  half_t* imp_w2_16 = (half_t*)ALLOC((size_t)512 * 512 * 2);
  half_t* ev_w_in16 = (half_t*)ALLOC((size_t)1536 * 512 * 2);
  half_t* ev_w_out16= (half_t*)ALLOC((size_t)512 * 512 * 2);
  half_t* w_ip16    = (half_t*)ALLOC((size_t)512 * 128 * 2);
  half_t* w_ig16    = (half_t*)ALLOC((size_t)512 * 1024 * 2);
  half_t* w_eg16    = (half_t*)ALLOC((size_t)512 * 1024 * 2);
  half_t* w_p1_16   = (half_t*)ALLOC((size_t)256 * 512 * 2);
  float*  q_ws      = (float*) ALLOC((size_t)32 * 512 * 4);
  float*  gates_ws  = (float*) ALLOC((size_t)32 * 2048 * 4);
  float*  h0_ws     = (float*) ALLOC((size_t)32 * 512 * 4);
  float*  gi_ws     = (float*) ALLOC((size_t)32 * 512 * 4);
  float*  ev_ws     = (float*) ALLOC((size_t)32 * 512 * 4);
  float*  h1_ws     = (float*) ALLOC((size_t)32 * 512 * 4);
  float*  ge_ws     = (float*) ALLOC((size_t)32 * 512 * 4);
  float*  p1_ws     = (float*) ALLOC((size_t)32 * 256 * 4);
  float*  c_ws      = (float*) ALLOC((size_t)32 * 512 * 4);
  half_t* x016g     = (half_t*)ALLOC((size_t)32 * 512 * 2);
  half_t* y16g      = (half_t*)ALLOC((size_t)32 * 512 * 2);

  auto cvt = [&](const float* src, half_t* dst, int n) {
    cvt_kernel<<<(n + 255) / 256, 256, 0, stream>>>(src, dst, n);
  };
  cvt(x, x16, BT * 128);
  cvt(w_ih, w_ih16, 2048 * 128);
  cvt(w_hh, w_hh16, 2048 * 512);
  cvt(imp_w1, imp_w1_16, 512 * 128);
  cvt(imp_w2, imp_w2_16, 512 * 512);
  cvt(ev_w_in, ev_w_in16, 1536 * 512);
  cvt(ev_w_out, ev_w_out16, 512 * 512);
  cvt(w_ip, w_ip16, 512 * 128);
  cvt(w_ig, w_ig16, 512 * 1024);
  cvt(w_eg, w_eg16, 512 * 1024);
  cvt(w_p1, w_p1_16, 256 * 512);

  lap_kernel<<<(BT * 128 + 255) / 256, 256, 0, stream>>>(x, lap16);
  gather_kernel<<<(BTN * 128 + 255) / 256, 256, 0, stream>>>(evs, embT, emb16);
  rowscale_kernel<<<(BTN + 255) / 256, 256, 0, stream>>>(td, rs, BTN);

  auto gemm = [&](const half_t* A, const half_t* W, const float* bias, const float* rscale,
                  float* oF, half_t* oH, int M, int N, int K, int act) {
    dim3 g(M / 128, N / 128, 1);
    gemm_f16_wmma<<<g, 256, 0, stream>>>(A, W, bias, rscale, oF, oH, M, N, K, act);
  };

  // imp MLP (time-parallel): imp1 = relu(emb@w1^T+b1); imp2u = imp1@w2^T+b2
  gemm(emb16,   imp_w1_16, imp_b1, nullptr, nullptr, imp1_16, BTN, 512, 128, 1);
  gemm(imp1_16, imp_w2_16, imp_b2, nullptr, nullptr, imp2_16, BTN, 512, 512, 0);
  // k/v: row-scale exp(-0.1 dt) commutes with the matmul; applied in epilogue.
  gemm(imp2_16, ev_w_in16 + (size_t)512 * 512,  ev_b_in + 512,  rs, nullptr, k16, BTN, 512, 512, 0);
  gemm(imp2_16, ev_w_in16 + (size_t)1024 * 512, ev_b_in + 1024, rs, nullptr, v16, BTN, 512, 512, 0);
  // ih = relu(lap@w_ip^T + b_ip) (f32 + f16); gates_x = x@w_ih^T + b_ih
  gemm(lap16, w_ip16, b_ip, nullptr, ih32, ih16, BT, 512, 128, 1);
  gemm(x16,   w_ih16, b_ih, nullptr, gatesx, nullptr, BT, 2048, 128, 0);

  scan_kernel<<<1, 512, 0, stream>>>(
      ev_w_in16 /*wq*/, ev_b_in /*bq*/,
      w_hh16, b_hh, gatesx,
      k16, v16,
      ev_w_out16, ev_b_out,
      w_ig16, b_ig, w_eg16, b_eg,
      w_p1_16, b_p1, w_p2, b_p2,
      ih16, ih32,
      q_ws, gates_ws, h0_ws, gi_ws, ev_ws, h1_ws, ge_ws, p1_ws, c_ws,
      x016g, y16g,
      (float*)d_out);
}
